// KipfNet_old_85014582657503
// MI455X (gfx1250) — matrix operand order repaired and checked by reading, not compiled, not run on hardware
//
#include <hip/hip_runtime.h>

typedef __attribute__((ext_vector_type(16))) _Float16 v16h;
typedef __attribute__((ext_vector_type(8)))  float    v8f;

#define N_FEAT 64
#define TB 256

// ---------------- elementwise / setup kernels ----------------

__global__ void zero_kernel(float* __restrict__ p, int n) {
    int i = blockIdx.x * blockDim.x + threadIdx.x;
    if (i < n) p[i] = 0.0f;
}

__global__ void neg_copy_kernel(float* __restrict__ dst, const float* __restrict__ src, int n) {
    int i = blockIdx.x * blockDim.x + threadIdx.x;
    if (i < n) dst[i] = -src[i];
}

__global__ void deg_kernel(const int* __restrict__ src, float* __restrict__ deg, int E) {
    int e = blockIdx.x * blockDim.x + threadIdx.x;
    if (e < E) {
        (void)__hip_atomic_fetch_add(&deg[src[e]], 1.0f,
                                     __ATOMIC_RELAXED, __HIP_MEMORY_SCOPE_AGENT);
    }
}

__global__ void dis_kernel(float* __restrict__ deg, int N) {
    int i = blockIdx.x * blockDim.x + threadIdx.x;
    if (i < N) {
        float d = deg[i];
        deg[i] = (d > 0.0f) ? rsqrtf(d) : 0.0f;
    }
}

__global__ void norm_kernel(const int* __restrict__ src, const int* __restrict__ dst,
                            const float* __restrict__ dis, float* __restrict__ norm, int E) {
    int e = blockIdx.x * blockDim.x + threadIdx.x;
    if (e < E) norm[e] = -dis[src[e]] * dis[dst[e]];
}

// ---------------- weight packing into WMMA B-fragment order (f16) ----------------
// Layout: Wp[k][nt][kkstep][lane][i] ; lane l -> col nt*16+(l&15), kr = kkstep*32+(i<8 ? (l>>4)*8+i : 16+(l>>4)*8+(i-8))
// Cols >= NOUT are zero-filled (layer 2: NOUT=10 padded to 16).

__global__ void pack_w_kernel(const float* __restrict__ W, _Float16* __restrict__ Wp,
                              int K, int NOUT, int NTILES) {
    int t = blockIdx.x * blockDim.x + threadIdx.x;
    int total = K * NTILES * 2 * 32 * 16;
    if (t >= total) return;
    int i      = t & 15;
    int lane   = (t >> 4) & 31;
    int kkstep = (t >> 9) & 1;
    int nt     = (t >> 10) % NTILES;
    int k      = (t >> 10) / NTILES;
    int lm = lane & 15, kh = lane >> 4;
    int kr  = kkstep * 32 + (i < 8 ? kh * 8 + i : 16 + kh * 8 + (i - 8));     // 0..63
    int col = nt * 16 + lm;
    float w = (col < NOUT) ? W[(k * N_FEAT + kr) * NOUT + col] : 0.0f;
    Wp[t] = (_Float16)w;
}

// ---------------- sparse propagation: out[dst] += scale*norm[e]*h[src] ----------------
// 16 threads per edge, float4 per thread (64 feats). L2-resident gather + f32 atomics.

__global__ void edge_prop_kernel(const float* __restrict__ h,
                                 const int* __restrict__ src, const int* __restrict__ dst,
                                 const float* __restrict__ norm,
                                 float* __restrict__ out, float scale, int E) {
    int t = blockIdx.x * blockDim.x + threadIdx.x;
    int e = t >> 4;
    if (e >= E) return;
    int c = (t & 15) << 2;
    float nv = scale * norm[e];
    const float4 v = *(const float4*)(h + (long)src[e] * N_FEAT + c);
    float* o = out + (long)dst[e] * N_FEAT + c;
    (void)__hip_atomic_fetch_add(o + 0, nv * v.x, __ATOMIC_RELAXED, __HIP_MEMORY_SCOPE_AGENT);
    (void)__hip_atomic_fetch_add(o + 1, nv * v.y, __ATOMIC_RELAXED, __HIP_MEMORY_SCOPE_AGENT);
    (void)__hip_atomic_fetch_add(o + 2, nv * v.z, __ATOMIC_RELAXED, __HIP_MEMORY_SCOPE_AGENT);
    (void)__hip_atomic_fetch_add(o + 3, nv * v.w, __ATOMIC_RELAXED, __HIP_MEMORY_SCOPE_AGENT);
}

// ---------------- WMMA GEMM: acc(M x NTILES_N*16) += A(M x 64) * Wp(k slice) ----------------
// 4 waves/block, one 16x16 tile per wave; K=64 as two v_wmma_f32_16x16x32_f16 steps.
// Branch-free: B fragments are pre-packed contiguous 32B per lane; tiles always full (M%16==0).

template<int NTILES_N>
__global__ __launch_bounds__(128) void
wmma_gemm_acc_kernel(const float* __restrict__ A,
                     const _Float16* __restrict__ Wp,   // packed slice for this k
                     float* __restrict__ acc, int M) {
    const int lane = threadIdx.x & 31;
    const int wave = threadIdx.x >> 5;
    const int m0 = (blockIdx.x * 4 + wave) * 16;
    if (m0 >= M) return;                      // wave-uniform guard
    const int nt = blockIdx.y;
    const int lm = lane & 15;
    const int kh = lane >> 4;
    const int STRIDE = NTILES_N * 16;

    // C tile: lane -> col nt*16+lm ; vgpr v -> row m0 + v + 8*kh
    float* accp = acc + (long)(m0 + 8 * kh) * STRIDE + nt * 16 + lm;
    v8f c;
    #pragma unroll
    for (int v = 0; v < 8; ++v) c[v] = accp[(long)v * STRIDE];

    const float* arow = A + (long)(m0 + lm) * N_FEAT;
    const _Float16* wp = Wp + (((long)nt * 2) * 32 + lane) * 16;

    #pragma unroll
    for (int kk = 0; kk < 64; kk += 32) {
        // A fragment: two 128-bit loads per lane half
        const float4* p0 = (const float4*)(arow + kk + kh * 8);
        const float4* p1 = (const float4*)(arow + kk + 16 + kh * 8);
        float4 f0 = p0[0], f1 = p0[1], f2 = p1[0], f3 = p1[1];
        v16h a;
        a[0]  = (_Float16)f0.x; a[1]  = (_Float16)f0.y; a[2]  = (_Float16)f0.z; a[3]  = (_Float16)f0.w;
        a[4]  = (_Float16)f1.x; a[5]  = (_Float16)f1.y; a[6]  = (_Float16)f1.z; a[7]  = (_Float16)f1.w;
        a[8]  = (_Float16)f2.x; a[9]  = (_Float16)f2.y; a[10] = (_Float16)f2.z; a[11] = (_Float16)f2.w;
        a[12] = (_Float16)f3.x; a[13] = (_Float16)f3.y; a[14] = (_Float16)f3.z; a[15] = (_Float16)f3.w;

        // B fragment: one contiguous 32B load per lane (packed)
        v16h b = *(const v16h*)(wp + (kk == 0 ? 0 : 512));

        c = __builtin_amdgcn_wmma_f32_16x16x32_f16(
                /*neg_a=*/false, a, /*neg_b=*/false, b,
                /*c_mod=*/(short)0, c, /*reuse_a=*/false, /*reuse_b=*/false);
    }

    #pragma unroll
    for (int v = 0; v < 8; ++v) accp[(long)v * STRIDE] = c[v];
}

// ---------------- epilogues ----------------

__global__ void relu_bias_kernel(const float* __restrict__ acc, const float* __restrict__ b,
                                 float* __restrict__ h, int n) {
    int i = blockIdx.x * blockDim.x + threadIdx.x;
    if (i < n) {
        float v = acc[i] + b[i & (N_FEAT - 1)];
        h[i] = v > 0.0f ? v : 0.0f;
    }
}

__global__ void out_bias_kernel(const float* __restrict__ acc, const float* __restrict__ b,
                                float* __restrict__ out, int n10) {
    int t = blockIdx.x * blockDim.x + threadIdx.x;
    if (t < n10) {
        int i = t / 10, f = t - i * 10;
        out[t] = acc[(long)i * 16 + f] + b[f];
    }
}

// ---------------- host orchestration ----------------

static inline unsigned nblk(long n) { return (unsigned)((n + TB - 1) / TB); }

extern "C" void kernel_launch(void* const* d_in, const int* in_sizes, int n_in,
                              void* d_out, int out_size, void* d_ws, size_t ws_size,
                              hipStream_t stream) {
    const float* x  = (const float*)d_in[0];
    const int* eidx = (const int*)d_in[1];
    const float* W1 = (const float*)d_in[2];
    const float* b1 = (const float*)d_in[3];
    const float* W2 = (const float*)d_in[4];
    const float* b2 = (const float*)d_in[5];
    float* out = (float*)d_out;

    const int N = in_sizes[0] / N_FEAT;     // 50000 (multiple of 16)
    const int E = in_sizes[1] / 2;          // 800000
    const int* esrc = eidx;
    const int* edst = eidx + E;

    // workspace carve-up (all 32B-aligned)
    float* ws   = (float*)d_ws;
    float* dis  = ws;                                   // N (rounded)
    float* norm = dis + ((N + 63) & ~63);               // E
    float* T0   = norm + ((E + 63) & ~63);              // N*64 each
    const int FN = N * N_FEAT;
    float* T1   = T0 + FN;
    float* T2   = T1 + FN;
    float* acc  = T2 + FN;                              // N*64 (layer1) / N*16 (layer2)
    _Float16* Wp1 = (_Float16*)(acc + FN);              // 8*4*2*32*16 = 32768 f16
    _Float16* Wp2 = Wp1 + 8 * 4 * 1024;                 // 8*1*2*32*16 = 8192 f16
    const int WSLICE1 = 4 * 1024;                       // f16 per k, layer 1
    const int WSLICE2 = 1 * 1024;                       // f16 per k, layer 2

    const int MT  = (N + 15) / 16;
    const int MB  = (MT + 3) / 4;                       // 4 waves per block
    const long ET = (long)E * 16;

    // ---- pack weights into WMMA B-fragment layout ----
    pack_w_kernel<<<nblk(8 * 4 * 1024), TB, 0, stream>>>(W1, Wp1, 8, 64, 4);
    pack_w_kernel<<<nblk(8 * 1 * 1024), TB, 0, stream>>>(W2, Wp2, 8, 10, 1);

    // ---- normalization: deg -> d^{-1/2} -> norm[e] = -dis[src]*dis[dst] ----
    zero_kernel<<<nblk(N), TB, 0, stream>>>(dis, N);
    deg_kernel<<<nblk(E), TB, 0, stream>>>(esrc, dis, E);
    dis_kernel<<<nblk(N), TB, 0, stream>>>(dis, N);
    norm_kernel<<<nblk(E), TB, 0, stream>>>(esrc, edst, dis, norm, E);

    // ================= layer 1: 64 -> 64, K=8 =================
    zero_kernel<<<nblk(FN), TB, 0, stream>>>(acc, FN);
    wmma_gemm_acc_kernel<4><<<dim3(MB, 4), 128, 0, stream>>>(x, Wp1, acc, N);

    zero_kernel<<<nblk(FN), TB, 0, stream>>>(T1, FN);
    edge_prop_kernel<<<nblk(ET), TB, 0, stream>>>(x, esrc, edst, norm, T1, 1.0f, E);
    wmma_gemm_acc_kernel<4><<<dim3(MB, 4), 128, 0, stream>>>(T1, Wp1 + WSLICE1, acc, N);

    {
        const float* tx0 = x;
        float* tx1 = T1;
        float* spares[2] = {T0, T2};
        int nsp = 2;
        for (int k = 2; k < 8; ++k) {
            float* tx2 = spares[--nsp];
            // Tx2 = 2 * prop(Tx1) - Tx0  (init -Tx0, scatter 2*norm*Tx1[src])
            neg_copy_kernel<<<nblk(FN), TB, 0, stream>>>(tx2, tx0, FN);
            edge_prop_kernel<<<nblk(ET), TB, 0, stream>>>(tx1, esrc, edst, norm, tx2, 2.0f, E);
            wmma_gemm_acc_kernel<4><<<dim3(MB, 4), 128, 0, stream>>>(tx2, Wp1 + (long)k * WSLICE1, acc, N);
            if (tx0 != x) spares[nsp++] = (float*)tx0;
            tx0 = tx1; tx1 = tx2;
        }
        // h = relu(acc + b1) into the remaining spare buffer
        float* h = spares[0];
        relu_bias_kernel<<<nblk(FN), TB, 0, stream>>>(acc, b1, h, FN);

        // ================= layer 2: 64 -> 10 (padded to 16), K=8 =================
        const int AN = N * 16;
        zero_kernel<<<nblk(AN), TB, 0, stream>>>(acc, AN);
        wmma_gemm_acc_kernel<1><<<dim3(MB, 1), 128, 0, stream>>>(h, Wp2, acc, N);

        float* u1 = (h == T1) ? T0 : T1;   // a buffer distinct from h
        zero_kernel<<<nblk(FN), TB, 0, stream>>>(u1, FN);
        edge_prop_kernel<<<nblk(ET), TB, 0, stream>>>(h, esrc, edst, norm, u1, 1.0f, E);
        wmma_gemm_acc_kernel<1><<<dim3(MB, 1), 128, 0, stream>>>(u1, Wp2 + WSLICE2, acc, N);

        // remaining third buffer
        float* u2 = (h != T0 && u1 != T0) ? T0 : ((h != T1 && u1 != T1) ? T1 : T2);
        const float* tx0b = h;
        float* tx1b = u1;
        float* sp2[2] = {u2, nullptr};
        int ns2 = 1;
        for (int k = 2; k < 8; ++k) {
            float* tx2 = sp2[--ns2];
            neg_copy_kernel<<<nblk(FN), TB, 0, stream>>>(tx2, tx0b, FN);
            edge_prop_kernel<<<nblk(ET), TB, 0, stream>>>(tx1b, esrc, edst, norm, tx2, 2.0f, E);
            wmma_gemm_acc_kernel<1><<<dim3(MB, 1), 128, 0, stream>>>(tx2, Wp2 + (long)k * WSLICE2, acc, N);
            sp2[ns2++] = (float*)tx0b;
            tx0b = tx1b; tx1b = tx2;
        }
        out_bias_kernel<<<nblk((long)N * 10), TB, 0, stream>>>(acc, b2, out, N * 10);
    }
}
